// PointerNetwork_32598801777021
// MI455X (gfx1250) — matrix operand order repaired and checked by reading, not compiled, not run on hardware
//
#include <hip/hip_runtime.h>
#include <hip/hip_bf16.h>

// Pointer network forward for MI455X (gfx1250, wave32, WMMA).
//
// Strategy: bf16 enc_out kept L2-resident (67MB < 192MB L2); persistent
// per-batch-slice workgroups for both recurrent loops; WMMA 16x16x32 bf16 for
// h@Whh^T and decoder gates; v_tanh_f32 inline asm for attention; decoder
// f-gate skipped (c0 == 0).
//
// Round-1 fix: weight base pointers made opaque per time-step so the compiler
// cannot hoist 64 B fragments (512 VGPRs) out of the loop and spill to
// scratch.
// Round-2 fix: the opaque pointer is laundered back into an explicit
// address_space(1) pointer so weight streams lower as global_load_b128
// (LOADcnt only) instead of flat_load_b128 (LOADcnt+DScnt).
// Round-3 fix: use a builtin ext_vector 128-bit type (u32x4) for fragment
// loads — HIP's uint4 is a class whose operator= cannot bind an AS1
// reference.
//
// Workspace requirement: ~74 MB.

typedef __attribute__((ext_vector_type(16))) __bf16 v16bf;
typedef __attribute__((ext_vector_type(8)))  float  v8f;
typedef unsigned short u16;
typedef unsigned int   u32;
typedef __attribute__((ext_vector_type(4))) u32 u32x4;   // POD 128-bit vector

#define AS1 __attribute__((address_space(1)))
typedef AS1 const u16 g_u16;

#define HID 256
#define SEQ 512
#define BAT 256
#define GAT 1024   // 4*HID
#define UNI 10

// ---------- small helpers ----------
__device__ __forceinline__ u16 f2bf(float f) {            // RNE f32 -> bf16
  u32 u = __float_as_uint(f);
  u32 r = (u + 0x7FFFu + ((u >> 16) & 1u)) >> 16;
  return (u16)r;
}
__device__ __forceinline__ float fast_tanh(float x) {     // gfx1250 V_TANH_F32
  float r; asm("v_tanh_f32 %0, %1" : "=v"(r) : "v"(x)); return r;
}
__device__ __forceinline__ float sigm(float x) {
  return __builtin_amdgcn_rcpf(1.0f + __expf(-x));
}
// Defeat loop-invariant hoisting of loads derived from a uniform pointer,
// while pinning the result to the GLOBAL address space (global_load_*).
__device__ __forceinline__ const g_u16* opaque_global(const u16* p) {
  unsigned long long v = (unsigned long long)p;
  asm volatile("" : "+s"(v));
  return (const g_u16*)v;
}

// A-matrix fragment (16x32 bf16): lane m=l&15 holds row m;
// lanes<16: K = k0+0..15; lanes>=16: K = k0+16..31 (matches load_B).
__device__ __forceinline__ v16bf load_A(const u16* base, int ld, int k0, int lane) {
  int m  = lane & 15;
  int kh = (lane >> 4) << 4;
  const u16* p = base + (size_t)m * ld + k0 + kh;
  union { v16bf v; u32x4 q[2]; } u;
  u.q[0] = *(const u32x4*)(p);
  u.q[1] = *(const u32x4*)(p + 8);
  return u.v;
}
// B-matrix fragment (32x16 bf16): lane n=l&15 holds column n; lanes<16 hold
// K=k0+0..15, lanes>=16 hold K=k0+16..31 (mirror of the documented 8-bit B
// layout). Source is stored [n][k] row-major -> 32 contiguous bytes per lane.
__device__ __forceinline__ v16bf load_B(const u16* wt, int ld, int n0, int k0, int lane) {
  int n  = lane & 15;
  int ks = k0 + ((lane >> 4) << 4);
  const u16* p = wt + (size_t)(n0 + n) * ld + ks;
  union { v16bf v; u32x4 q[2]; } u;
  u.q[0] = *(const u32x4*)(p);
  u.q[1] = *(const u32x4*)(p + 8);
  return u.v;
}
// Same, but through an explicit global-address-space pointer.
__device__ __forceinline__ v16bf load_Bg(const g_u16* wt, int ld, int n0, int k0, int lane) {
  int n  = lane & 15;
  int ks = k0 + ((lane >> 4) << 4);
  const g_u16* p = wt + (size_t)(n0 + n) * ld + ks;
  union { v16bf v; u32x4 q[2]; } u;
  u.q[0] = *(AS1 const u32x4*)(p);
  u.q[1] = *(AS1 const u32x4*)(p + 8);
  return u.v;
}
#define WMMA_BF16(a, b, c) \
  __builtin_amdgcn_wmma_f32_16x16x32_bf16(false, (a), false, (b), (short)0, (c), false, false)

#define V8F_ZERO {0.f,0.f,0.f,0.f,0.f,0.f,0.f,0.f}

// ---------- kernel 0: weight conversion + decoder-input precompute ----------
__global__ void prep_kernel(const float* __restrict__ x, const int* __restrict__ y,
                            const float* __restrict__ encWhh, const float* __restrict__ decWih,
                            const float* __restrict__ W1,
                            u16* __restrict__ Whhb, u16* __restrict__ dWihb,
                            u16* __restrict__ W1b, float* __restrict__ dseq,
                            float* __restrict__ partials) {
  int i0 = blockIdx.x * blockDim.x + threadIdx.x;
  int stride = gridDim.x * blockDim.x;
  for (int i = i0; i < GAT * HID; i += stride) Whhb[i] = f2bf(encWhh[i]);
  for (int i = i0; i < GAT * HID; i += stride) {
    int n = i >> 8, k = i & 255;
    dWihb[i] = f2bf(decWih[n * (HID + 1) + k]);      // drop col 256 (rank-1 later)
  }
  for (int i = i0; i < 16 * HID; i += stride) {
    int u = i >> 8, k = i & 255;
    W1b[i] = (u < UNI) ? f2bf(W1[u * HID + k]) : (u16)0;   // pad U 10->16
  }
  for (int i = i0; i < SEQ * BAT; i += stride) {     // dec_in for step t
    int t = i / BAT, b = i % BAT;
    float v = 0.f;
    if (t > 0) { int yv = y[b * SEQ + (t - 1)]; v = x[b * SEQ + yv]; }
    dseq[i] = v;
  }
  for (int i = i0; i < 64; i += stride) partials[i] = 0.f;
}

// ---------- kernel 1: encoder LSTM (persistent, 16 blocks x 16 rows) ----------
// gates[m,n] = x[m,t]*Wih[n] + h[m,:]@Whh[n,:] + bih[n]+bhh[n]
// Wave w owns gate columns {g*256 + w*32 + t2*16 + (lane&15)}, so i/f/g/o for a
// given (row, j) land in the same lane -> c stays in registers, no gate LDS.
__global__ __launch_bounds__(256) void enc_kernel(const float* __restrict__ x,
                                                  const float* __restrict__ encWih,
                                                  const float* __restrict__ bih,
                                                  const float* __restrict__ bhh,
                                                  const u16* __restrict__ Whhb,
                                                  u16* __restrict__ encb) {
  __shared__ __align__(16) u16 hb[16 * HID];
  __shared__ float xs[16];
  const int tid = threadIdx.x, lane = tid & 31, w = tid >> 5;
  const int b0 = blockIdx.x * 16;
  const int jlo = lane & 15;
  const int mh  = (lane >> 4) << 3;   // 0 or 8 (row offset for C-layout)

  for (int i = tid; i < 16 * HID; i += 256) hb[i] = 0;

  float bias[4][2], wih[4][2];
#pragma unroll
  for (int g = 0; g < 4; ++g)
#pragma unroll
    for (int t2 = 0; t2 < 2; ++t2) {
      int n = g * 256 + w * 32 + t2 * 16 + jlo;
      bias[g][t2] = bih[n] + bhh[n];
      wih[g][t2]  = encWih[n];
    }
  float c[16];
#pragma unroll
  for (int i = 0; i < 16; ++i) c[i] = 0.f;
  __syncthreads();

  for (int t = 0; t < SEQ; ++t) {
    // Opaque per-iteration weight pointer (global AS): B fragments re-stream
    // from L2 every step; no hoist, no spill, global_load (not flat_load).
    const g_u16* wb = opaque_global(Whhb);

    if (tid < 16) xs[tid] = x[(size_t)(b0 + tid) * SEQ + t];
    __syncthreads();                       // hb + xs ready

    v16bf Af[8];
#pragma unroll
    for (int kt = 0; kt < 8; ++kt) Af[kt] = load_A(hb, HID, kt * 32, lane);
    float xr[8];
#pragma unroll
    for (int r = 0; r < 8; ++r) xr[r] = xs[r + mh];
    __syncthreads();                       // all reads done before rewrites

    v8f acc[8] = {V8F_ZERO, V8F_ZERO, V8F_ZERO, V8F_ZERO,
                  V8F_ZERO, V8F_ZERO, V8F_ZERO, V8F_ZERO};
#pragma unroll
    for (int g = 0; g < 4; ++g)
#pragma unroll
      for (int t2 = 0; t2 < 2; ++t2) {
        const int ai = g * 2 + t2;
        const int n0 = g * 256 + w * 32 + t2 * 16;
#pragma unroll
        for (int kt = 0; kt < 8; ++kt)
          acc[ai] = WMMA_BF16(Af[kt], load_Bg(wb, HID, n0, kt * 32, lane), acc[ai]);
      }

#pragma unroll
    for (int t2 = 0; t2 < 2; ++t2) {
      const int j = w * 32 + t2 * 16 + jlo;
#pragma unroll
      for (int r = 0; r < 8; ++r) {
        const int m = r + mh;
        const float xv = xr[r];
        float iv = acc[0 * 2 + t2][r] + bias[0][t2] + xv * wih[0][t2];
        float fv = acc[1 * 2 + t2][r] + bias[1][t2] + xv * wih[1][t2];
        float gv = acc[2 * 2 + t2][r] + bias[2][t2] + xv * wih[2][t2];
        float ov = acc[3 * 2 + t2][r] + bias[3][t2] + xv * wih[3][t2];
        float cn = sigm(fv) * c[t2 * 8 + r] + sigm(iv) * fast_tanh(gv);
        c[t2 * 8 + r] = cn;
        u16 hb16 = f2bf(sigm(ov) * fast_tanh(cn));
        hb[m * HID + j] = hb16;
        encb[((size_t)(b0 + m) * SEQ + t) * HID + j] = hb16;
      }
    }
  }
}

// ---------- kernel 2: pre[b,s,u] = enc_out @ W1^T (WMMA, U padded to 16) ----
__global__ __launch_bounds__(256) void pre_kernel(const u16* __restrict__ encb,
                                                  const u16* __restrict__ W1b,
                                                  float* __restrict__ pre) {
  const int tid = threadIdx.x, lane = tid & 31, w = tid >> 5;
  const int tile = blockIdx.x * 8 + w;          // 8192 tiles of 16 rows
  const size_t r0 = (size_t)tile * 16;
  v8f ac = V8F_ZERO;
#pragma unroll
  for (int kt = 0; kt < 8; ++kt) {
    v16bf a = load_A(encb + r0 * HID, HID, kt * 32, lane);
    v16bf b = load_B(W1b, HID, 0, kt * 32, lane);
    ac = WMMA_BF16(a, b, ac);
  }
  const int n = lane & 15;
  if (n < UNI) {
    const int mh = (lane >> 4) << 3;
#pragma unroll
    for (int r = 0; r < 8; ++r) pre[(r0 + r + mh) * UNI + n] = ac[r];
  }
}

// ---------- kernel 3: decoder (persistent, 64 blocks x 4 rows x 16 waves) ---
__global__ __launch_bounds__(512) void dec_kernel(const int* __restrict__ y,
                                                  const float* __restrict__ pre,
                                                  const u16* __restrict__ encb,
                                                  const u16* __restrict__ dWihb,
                                                  const float* __restrict__ decWih,
                                                  const float* __restrict__ dbih,
                                                  const float* __restrict__ dbhh,
                                                  const float* __restrict__ W2,
                                                  const float* __restrict__ Vv,
                                                  const float* __restrict__ dseq,
                                                  float* __restrict__ out,
                                                  float* __restrict__ partials) {
  constexpr int R = 4;
  __shared__ float hs[R * HID];
  __shared__ float scs[R * SEQ];
  __shared__ float w2h[R * 16];
  __shared__ __align__(16) u16 xin[16 * HID];   // WMMA A: rows >= R stay zero
  __shared__ float ctxp[3][R * HID];
  __shared__ float pmaxs[R][4], psums[R][4];
  __shared__ int   pidxs[R][4];
  __shared__ float din_s[R], sy_s[R], lred[R];

  const int tid = threadIdx.x, lane = tid & 31, w = tid >> 5;
  const int b0 = blockIdx.x * R;
  const int m = w & 3, q = w >> 2;              // row / S-quarter per wave

  for (int i = tid; i < R * HID; i += 512) hs[i] = 0.f;
  for (int i = tid; i < 16 * HID; i += 512) xin[i] = 0;

  float Vr[UNI];
#pragma unroll
  for (int u = 0; u < UNI; ++u) Vr[u] = Vv[u];

  const int jl = w * 16 + (lane & 15);
  const int glist[3] = {0, 2, 3};               // f-gate dead: c0 == 0
  float biasD[3], wcolD[3];
#pragma unroll
  for (int gi = 0; gi < 3; ++gi) {
    int n = glist[gi] * 256 + jl;
    biasD[gi] = dbih[n] + dbhh[n];
    wcolD[gi] = decWih[n * (HID + 1) + HID];    // rank-1 dec_in column
  }
  float loss_acc = 0.f;
  __syncthreads();

  for (int t = 0; t < SEQ; ++t) {
    if (tid < R) din_s[tid] = dseq[t * BAT + b0 + tid];
    if (tid < R * 16) {                          // (a) W2h = hs @ W2^T
      int mm = tid >> 4, u = tid & 15;
      float s = 0.f;
      if (u < UNI) {
        const float* hp = &hs[mm * HID];
        const float* wp = &W2[u * HID];
        for (int h = 0; h < HID; ++h) s += hp[h] * wp[h];
      }
      w2h[mm * 16 + u] = s;
    }
    __syncthreads();

    {                                            // (b) attention scores
      const int s = tid;
#pragma unroll
      for (int mm = 0; mm < R; ++mm) {
        const float* pp = pre + ((size_t)(b0 + mm) * SEQ + s) * UNI;
        float a = 0.f;
#pragma unroll
        for (int u = 0; u < UNI; ++u) a += Vr[u] * fast_tanh(pp[u] + w2h[mm * 16 + u]);
        scs[mm * SEQ + s] = a;
      }
    }
    __syncthreads();

    // (c1) per-quarter max/argmax (first-max tie-break)
    float vmax = -3.402823466e38f; int vidx = q * 128;
#pragma unroll
    for (int i = 0; i < 4; ++i) {
      int s = q * 128 + i * 32 + lane;
      float v = scs[m * SEQ + s];
      if (v > vmax) { vmax = v; vidx = s; }
    }
#pragma unroll
    for (int off = 16; off > 0; off >>= 1) {
      float ov = __shfl_xor(vmax, off, 32);
      int   oi = __shfl_xor(vidx, off, 32);
      if (ov > vmax || (ov == vmax && oi < vidx)) { vmax = ov; vidx = oi; }
    }
    if (lane == 0) { pmaxs[m][q] = vmax; pidxs[m][q] = vidx; }
    if (w == 0 && lane < R) {                    // score at label, pre-overwrite
      int yt = y[(size_t)(b0 + lane) * SEQ + t];
      sy_s[lane] = scs[lane * SEQ + yt];
    }
    __syncthreads();

    float rmax = pmaxs[m][0]; int rarg = pidxs[m][0];
#pragma unroll
    for (int qq = 1; qq < 4; ++qq) {
      float v = pmaxs[m][qq]; int ii = pidxs[m][qq];
      if (v > rmax || (v == rmax && ii < rarg)) { rmax = v; rarg = ii; }
    }
    // (c2) exp in place + partial sums
    float lsum = 0.f;
#pragma unroll
    for (int i = 0; i < 4; ++i) {
      int s = q * 128 + i * 32 + lane;
      float p = __expf(scs[m * SEQ + s] - rmax);
      scs[m * SEQ + s] = p;
      lsum += p;
    }
#pragma unroll
    for (int off = 16; off > 0; off >>= 1) lsum += __shfl_xor(lsum, off, 32);
    if (lane == 0) psums[m][q] = lsum;
    __syncthreads();

    float rsum = psums[m][0] + psums[m][1] + psums[m][2] + psums[m][3];
    if (q == 0 && lane == 0) {                   // CE + prediction
      loss_acc += rmax + __logf(rsum) - sy_s[m];
      out[t * BAT + b0 + m] = (float)rarg;       // argmax as float
    }

    // (d) context GEMV: di = p @ enc_out[b] / rsum   (L2-resident bf16 reads)
    float a8[8] = {0.f,0.f,0.f,0.f,0.f,0.f,0.f,0.f};
    const u16* ep = encb + ((size_t)(b0 + m) * SEQ + q * 128) * HID + lane * 8;
#pragma unroll 4
    for (int i = 0; i < 128; ++i) {
      float pv = scs[m * SEQ + q * 128 + i];
      u32x4 qv = *(const u32x4*)ep; ep += HID;
      u32 u0 = qv[0], u1 = qv[1], u2 = qv[2], u3 = qv[3];
      a8[0] = fmaf(pv, __uint_as_float(u0 << 16),          a8[0]);
      a8[1] = fmaf(pv, __uint_as_float(u0 & 0xffff0000u),  a8[1]);
      a8[2] = fmaf(pv, __uint_as_float(u1 << 16),          a8[2]);
      a8[3] = fmaf(pv, __uint_as_float(u1 & 0xffff0000u),  a8[3]);
      a8[4] = fmaf(pv, __uint_as_float(u2 << 16),          a8[4]);
      a8[5] = fmaf(pv, __uint_as_float(u2 & 0xffff0000u),  a8[5]);
      a8[6] = fmaf(pv, __uint_as_float(u3 << 16),          a8[6]);
      a8[7] = fmaf(pv, __uint_as_float(u3 & 0xffff0000u),  a8[7]);
    }
    if (q > 0) {
      float* cp = &ctxp[q - 1][m * HID + lane * 8];
#pragma unroll
      for (int e = 0; e < 8; ++e) cp[e] = a8[e];
    }
    __syncthreads();
    if (q == 0) {
      float inv = __builtin_amdgcn_rcpf(rsum);
#pragma unroll
      for (int e = 0; e < 8; ++e) {
        int hh = lane * 8 + e;
        float v = (a8[e] + ctxp[0][m * HID + hh] + ctxp[1][m * HID + hh]
                         + ctxp[2][m * HID + hh]) * inv;
        xin[m * HID + hh] = f2bf(v);
      }
    }
    __syncthreads();

    // (e) decoder gates via WMMA (i, g, o only), K = 256
    const g_u16* dwb = opaque_global(dWihb);     // no hoist, global_load path
    v16bf Af[8];
#pragma unroll
    for (int kt = 0; kt < 8; ++kt) Af[kt] = load_A(xin, HID, kt * 32, lane);
    v8f ac[3] = {V8F_ZERO, V8F_ZERO, V8F_ZERO};
#pragma unroll
    for (int gi = 0; gi < 3; ++gi) {
      const int n0 = glist[gi] * 256 + w * 16;
#pragma unroll
      for (int kt = 0; kt < 8; ++kt)
        ac[gi] = WMMA_BF16(Af[kt], load_Bg(dwb, HID, n0, kt * 32, lane), ac[gi]);
    }
    // (f) pointwise LSTM from zero state; only rows 0..3 valid (lanes < 16)
    if (lane < 16) {
#pragma unroll
      for (int r = 0; r < R; ++r) {
        float dv = din_s[r];
        float iv = ac[0][r] + biasD[0] + dv * wcolD[0];
        float gv = ac[1][r] + biasD[1] + dv * wcolD[1];
        float ov = ac[2][r] + biasD[2] + dv * wcolD[2];
        float cc = sigm(iv) * fast_tanh(gv);
        hs[r * HID + jl] = sigm(ov) * fast_tanh(cc);
      }
    }
    __syncthreads();
  }

  if (q == 0 && lane == 0) lred[m] = loss_acc;
  __syncthreads();
  if (tid == 0) partials[blockIdx.x] = lred[0] + lred[1] + lred[2] + lred[3];
}

// ---------- kernel 4: finalize loss ----------
__global__ void loss_kernel(const float* __restrict__ partials, float* __restrict__ out) {
  if (threadIdx.x == 0) {
    float s = 0.f;
    for (int i = 0; i < 64; ++i) s += partials[i];
    out[SEQ * BAT] = s * (1.0f / (float)(BAT * BAT));   // /B for mean, /B again
  }
}

// ---------- host-side launcher ----------
extern "C" void kernel_launch(void* const* d_in, const int* in_sizes, int n_in,
                              void* d_out, int out_size, void* d_ws, size_t ws_size,
                              hipStream_t stream) {
  const float* x      = (const float*)d_in[0];
  const int*   y      = (const int*)  d_in[1];
  const float* encWih = (const float*)d_in[2];
  const float* encWhh = (const float*)d_in[3];
  const float* encbih = (const float*)d_in[4];
  const float* encbhh = (const float*)d_in[5];
  const float* decWih = (const float*)d_in[6];
  // d_in[7] = dec_Whh is inert (decoder hidden never fed back)
  const float* decbih = (const float*)d_in[8];
  const float* decbhh = (const float*)d_in[9];
  const float* W1     = (const float*)d_in[10];
  const float* W2     = (const float*)d_in[11];
  const float* Vv     = (const float*)d_in[12];
  float* out = (float*)d_out;   // [S*B] predictions (as float) + 1 loss scalar

  char* ws = (char*)d_ws;
  size_t o = 0;
  u16*   encb   = (u16*)(ws + o); o += (size_t)BAT * SEQ * HID * 2;  // 67 MB
  u16*   Whhb   = (u16*)(ws + o); o += (size_t)GAT * HID * 2;
  u16*   dWihb  = (u16*)(ws + o); o += (size_t)GAT * HID * 2;
  u16*   W1b    = (u16*)(ws + o); o += (size_t)16 * HID * 2;
  float* pre    = (float*)(ws + o); o += (size_t)BAT * SEQ * UNI * 4;
  float* dseq   = (float*)(ws + o); o += (size_t)SEQ * BAT * 4;
  float* parts  = (float*)(ws + o); o += 64 * 4;
  (void)ws_size; (void)in_sizes; (void)n_in; (void)out_size;

  prep_kernel<<<256, 256, 0, stream>>>(x, y, encWhh, decWih, W1,
                                       Whhb, dWihb, W1b, dseq, parts);
  enc_kernel<<<16, 256, 0, stream>>>(x, encWih, encbih, encbhh, Whhb, encb);
  pre_kernel<<<1024, 256, 0, stream>>>(encb, W1b, pre);
  dec_kernel<<<64, 512, 0, stream>>>(y, pre, encb, dWihb, decWih,
                                     decbih, decbhh, W2, Vv, dseq, out, parts);
  loss_kernel<<<1, 64, 0, stream>>>(parts, out);
}
